// MemoryConsolidationManagerV2_16973710754182
// MI455X (gfx1250) — matrix operand order repaired and checked by reading, not compiled, not run on hardware
//
#include <hip/hip_runtime.h>
#include <hip/hip_bf16.h>

// ---------------------------------------------------------------------------
// MemoryConsolidationManagerV2 for MI455X (gfx1250, wave32, WMMA bf16 path)
//
//  K0 stage_weights : W1/W2 f32 -> bf16, pre-swizzled into WMMA B-fragment
//                     lane layout (ISA 7.12.2), written to workspace.
//  K1 topo_map      : per-wave 16-row strip, N=128 (8 n-tiles):
//                     TDM-copy staged W1 chunk (128KB) + W2 (32KB) into LDS,
//                     GEMM1 (K=1024) + bias + leakyrelu, H re-striped via LDS,
//                     GEMM2 (K=128) + bias + tanh -> topo[65536,128] f32.
//                     A-side global loads double-buffered; all 8 B-fragments
//                     kept live in distinct registers per k-step.
//  K2 centroid, K3 dists, K4 select (bitonic sort + masked mean).
// ---------------------------------------------------------------------------

#define B_ 32
#define T_ 2048
#define D_ 1024
#define P_ 128

typedef __bf16 bf16_t;
typedef __bf16 bf16x8  __attribute__((ext_vector_type(8)));
typedef __bf16 bf16x16 __attribute__((ext_vector_type(16)));
typedef float  f32x8   __attribute__((ext_vector_type(8)));
typedef unsigned int u32x4 __attribute__((ext_vector_type(4)));
typedef int    i32x4   __attribute__((ext_vector_type(4)));
typedef int    i32x8   __attribute__((ext_vector_type(8)));

// workspace byte offsets
#define OFF_W1S  0u          // staged W1 bf16: 2 chunks * 131072 B
#define OFF_W2S  262144u     // staged W2 bf16: 32768 B
#define OFF_CEN  294912u     // centroid f32: 32*128*4
#define OFF_DST  311296u     // dists f32: 32*2048*4
#define OFF_TOPO 573440u     // topo f32: 65536*128*4 = 32 MB

#define LDS_W2_OFF 131072
#define LDS_BYTES  163840    // 128KB W1 chunk + 32KB W2 (H tile reuses W1 region)

#if defined(__HIP_DEVICE_COMPILE__) && __has_builtin(__builtin_amdgcn_tensor_load_to_lds)
#define HAVE_TDM 1
#else
#define HAVE_TDM 0
#endif

// Copy `units8` 8-byte units from global (pre-staged bf16 blob) into LDS.
// TDM path: wave 0 issues a 1-row 2D tensor descriptor, waits TENSORcnt.
__device__ __forceinline__ void stage_to_lds(char* smem, const void* gsrc,
                                             unsigned lds_off, unsigned units8,
                                             int tid) {
#if HAVE_TDM
  if ((tid >> 5) == 0) {
    unsigned long long ga = (unsigned long long)gsrc;
    u32x4 g0;
    g0[0] = 1u;                                          // count=1 (valid D#)
    g0[1] = lds_off;                                     // lds_addr (bytes)
    g0[2] = (unsigned)(ga & 0xffffffffu);                // global_addr[31:0]
    g0[3] = (unsigned)((ga >> 32) & 0x1ffffffu) | (2u << 30); // addr[56:32] | type=2
    i32x8 g1;
    g1[0] = 0x00030000;                                  // data_size=3 (8B)
    g1[1] = (int)((units8 & 0xffffu) << 16);             // tensor_dim0 lo16
    g1[2] = (int)(((units8 >> 16) & 0xffffu) | (1u << 16)); // dim0 hi16 | tensor_dim1=1
    g1[3] = (int)((units8 & 0xffffu) << 16);             // tile_dim0
    g1[4] = 1;                                           // tile_dim1=1, tile_dim2=0
    g1[5] = (int)units8;                                 // tensor_dim0_stride lo32
    g1[6] = 0;
    g1[7] = 0;
    i32x4 z4 = {0, 0, 0, 0};
#if defined(__clang_major__) && __clang_major__ >= 23
    i32x8 z8 = {0, 0, 0, 0, 0, 0, 0, 0};
    __builtin_amdgcn_tensor_load_to_lds(g0, g1, z4, z4, z8, 0);
#else
    __builtin_amdgcn_tensor_load_to_lds(g0, g1, z4, z4, 0);
#endif
    __builtin_amdgcn_s_wait_tensorcnt(0);
  }
#else
  uint4* dst = (uint4*)(smem + lds_off);
  const uint4* src = (const uint4*)gsrc;
  const unsigned n16 = units8 >> 1;
  for (unsigned i = (unsigned)tid; i < n16; i += 256u) dst[i] = src[i];
#endif
}

// Load one A-fragment's 16 f32 (two 32B spans) into 4 float4 registers.
__device__ __forceinline__ void load_a(const float* __restrict__ arow, int kk,
                                       int sel, float4* a) {
  a[0] = *(const float4*)(arow + kk + sel);
  a[1] = *(const float4*)(arow + kk + sel + 4);
  a[2] = *(const float4*)(arow + kk + 16 + sel);
  a[3] = *(const float4*)(arow + kk + 20 + sel);
}

__device__ __forceinline__ bf16x16 cvt_a(const float4* a) {
  bf16x16 af;
  af[0] = (bf16_t)a[0].x; af[1] = (bf16_t)a[0].y;
  af[2] = (bf16_t)a[0].z; af[3] = (bf16_t)a[0].w;
  af[4] = (bf16_t)a[1].x; af[5] = (bf16_t)a[1].y;
  af[6] = (bf16_t)a[1].z; af[7] = (bf16_t)a[1].w;
  af[8] = (bf16_t)a[2].x; af[9] = (bf16_t)a[2].y;
  af[10] = (bf16_t)a[2].z; af[11] = (bf16_t)a[2].w;
  af[12] = (bf16_t)a[3].x; af[13] = (bf16_t)a[3].y;
  af[14] = (bf16_t)a[3].z; af[15] = (bf16_t)a[3].w;
  return af;
}

// One k-step across all 8 n-tiles: materialize all 8 B-fragments in distinct
// live registers first, then issue the 8 WMMAs (waits pipeline down, no WAR
// hazards on B operands, no dscnt-0 stalls on the tail tiles).
__device__ __forceinline__ void wmma_8(const bf16x8* __restrict__ wlds,
                                       int base, int jstride,
                                       const bf16x16 af, f32x8* acc) {
  bf16x16 bfm[8];
#pragma unroll
  for (int j = 0; j < 8; ++j) {
    bf16x8 lo = wlds[base + j * jstride];
    bf16x8 hi = wlds[base + j * jstride + 32];
    bfm[j] = __builtin_shufflevector(lo, hi,
        0, 1, 2, 3, 4, 5, 6, 7, 8, 9, 10, 11, 12, 13, 14, 15);
  }
#pragma unroll
  for (int j = 0; j < 8; ++j) {
    acc[j] = __builtin_amdgcn_wmma_f32_16x16x32_bf16(
        false, af, false, bfm[j], (short)0, acc[j], false, false);
  }
}

// ---------------------------------------------------------------------------
// K0: convert W1/W2 to bf16, pre-swizzled into B-fragment groups of 8 halves.
// ---------------------------------------------------------------------------
__global__ void __launch_bounds__(256) stage_weights_kernel(
    const float* __restrict__ W1, const float* __restrict__ W2,
    unsigned short* __restrict__ w1s_u, unsigned short* __restrict__ w2s_u) {
  const int g = blockIdx.x * 256 + threadIdx.x;
  if (g < 16384) {                       // W1: 2 chunks x 8192 groups
    const int c    = g >> 13;
    const int gi   = g & 8191;
    const int jsl  = gi >> 6;            // j*16 + sl  (0..127)
    const int part = (gi >> 5) & 1;
    const int lane = gi & 31;
    const int j    = jsl >> 4;
    const int sl   = jsl & 15;
    const int n    = j * 16 + (lane & 15);
    const int kb   = (c * 16 + sl) * 32 + ((lane & 16) ? 8 : 0) + part * 16;
    bf16_t* dst = (bf16_t*)w1s_u + (size_t)g * 8;
#pragma unroll
    for (int e = 0; e < 8; ++e) dst[e] = (bf16_t)W1[(size_t)(kb + e) * P_ + n];
  } else if (g < 16384 + 2048) {         // W2: 32 groups-of-64 (j*4 + s)
    const int g2   = g - 16384;
    const int js   = g2 >> 6;
    const int part = (g2 >> 5) & 1;
    const int lane = g2 & 31;
    const int j    = js >> 2;
    const int s    = js & 3;
    const int n    = j * 16 + (lane & 15);
    const int kb   = s * 32 + ((lane & 16) ? 8 : 0) + part * 16;
    bf16_t* dst = (bf16_t*)w2s_u + (size_t)g2 * 8;
#pragma unroll
    for (int e = 0; e < 8; ++e) dst[e] = (bf16_t)W2[(size_t)(kb + e) * P_ + n];
  }
}

// ---------------------------------------------------------------------------
// K1: fused topological map. 512 blocks x 256 threads (8 waves); each wave
// owns a 16-row strip x full N=128 (8 accumulator tiles, 64 VGPRs).
// LDS caps us at 1 block/WGP = 2 waves/SIMD, so declare it: min 2 waves/EU
// frees the full 512-VGPR budget for the live B-fragment array.
// ---------------------------------------------------------------------------
__global__ void __launch_bounds__(256, 2) topo_map_kernel(
    const float* __restrict__ mem, const float* __restrict__ b1,
    const float* __restrict__ b2, const unsigned short* __restrict__ w1s_u,
    const unsigned short* __restrict__ w2s_u, float* __restrict__ topo) {
  extern __shared__ char smem[];
  const int tid  = threadIdx.x;
  const int lane = tid & 31;
  const int wave = tid >> 5;
  const int l16  = lane & 15;
  const int sel  = (lane & 16) ? 8 : 0;   // A-fragment K sub-select
  const int mOff = (lane & 16) ? 8 : 0;   // C-layout row offset

  stage_to_lds(smem, w1s_u, 0, 16384, tid);            // W1 chunk 0 (128 KB)
  stage_to_lds(smem, w2s_u, LDS_W2_OFF, 4096, tid);    // W2 (32 KB)
  __syncthreads();

  const size_t R0 = (size_t)blockIdx.x * 128 + (size_t)wave * 16;
  const float* arow = mem + (R0 + (size_t)l16) * D_;

  const f32x8 zero = {0.f, 0.f, 0.f, 0.f, 0.f, 0.f, 0.f, 0.f};
  f32x8 acc[8];
#pragma unroll
  for (int j = 0; j < 8; ++j) acc[j] = zero;

  const bf16x8* wlds = (const bf16x8*)smem;

  // ---- GEMM1: K = 1024 in two 512-wide LDS-resident chunks -------------
#pragma unroll 1
  for (int c = 0; c < 2; ++c) {
    if (c == 1) {
      __syncthreads();                                  // done reading chunk 0
      stage_to_lds(smem, w1s_u + 65536, 0, 16384, tid); // W1 chunk 1
      __syncthreads();
    }
    const int k0 = c * 512;
    float4 aA[4], aB[4];
    load_a(arow, k0, sel, aA);                          // prologue: k-step 0
#pragma unroll 1
    for (int sp = 0; sp < 8; ++sp) {                    // 2 k-steps per iter
      load_a(arow, k0 + (2 * sp + 1) * 32, sel, aB);    // prefetch odd step
      wmma_8(wlds, (2 * sp) * 64 + lane, 1024, cvt_a(aA), acc);
      if (sp < 7)
        load_a(arow, k0 + (2 * sp + 2) * 32, sel, aA);  // prefetch next even
      wmma_8(wlds, (2 * sp + 1) * 64 + lane, 1024, cvt_a(aB), acc);
    }
  }

  __syncthreads();   // all waves done with W1 region; reuse it for H tiles

  // ---- bias + LeakyReLU; re-stripe C-layout -> A-layout via LDS --------
  bf16_t* H = (bf16_t*)smem + wave * 2176;   // 16 x 128 bf16, row stride 136
#pragma unroll
  for (int j = 0; j < 8; ++j) {
    const float bb = b1[j * 16 + l16];
#pragma unroll
    for (int i = 0; i < 8; ++i) {
      float x = acc[j][i] + bb;
      x = (x > 0.f) ? x : 0.01f * x;
      H[(i + mOff) * 136 + j * 16 + l16] = (bf16_t)x;
      acc[j][i] = 0.f;
    }
  }
  __syncthreads();

  // ---- GEMM2: K = 128, W2 resident in LDS ------------------------------
  const bf16x8* w2lds = (const bf16x8*)(smem + LDS_W2_OFF);
  const bf16_t* Hrow  = (const bf16_t*)smem + wave * 2176 + l16 * 136;
#pragma unroll
  for (int s2 = 0; s2 < 4; ++s2) {
    const int kk2 = s2 * 32;
    bf16x8 lo = *(const bf16x8*)(Hrow + kk2 + sel);
    bf16x8 hi = *(const bf16x8*)(Hrow + kk2 + 16 + sel);
    bf16x16 af = __builtin_shufflevector(lo, hi,
        0, 1, 2, 3, 4, 5, 6, 7, 8, 9, 10, 11, 12, 13, 14, 15);
    wmma_8(w2lds, s2 * 64 + lane, 256, af, acc);
  }

  // ---- bias + tanh, store topo -----------------------------------------
#pragma unroll
  for (int j = 0; j < 8; ++j) {
    const float bb = b2[j * 16 + l16];
#pragma unroll
    for (int i = 0; i < 8; ++i) {
      const float x = tanhf(acc[j][i] + bb);
      topo[(R0 + i + mOff) * P_ + j * 16 + l16] = x;
    }
  }
}

// ---------------------------------------------------------------------------
// K2: centroid[b][p] = mean_t topo[b][t][p] (deterministic serial sum).
// ---------------------------------------------------------------------------
__global__ void __launch_bounds__(128) centroid_kernel(
    const float* __restrict__ topo, float* __restrict__ cen) {
  const int b = blockIdx.x, p = threadIdx.x;
  const float* base = topo + (size_t)b * T_ * P_ + p;
  float s = 0.f;
  for (int t = 0; t < T_; ++t) s += base[(size_t)t * P_];
  cen[b * P_ + p] = s * (1.f / (float)T_);
}

// ---------------------------------------------------------------------------
// K3: one wave per token: dists[b][t] = ||topo[b][t] - centroid[b]||_2
// ---------------------------------------------------------------------------
__global__ void __launch_bounds__(256) dists_kernel(
    const float* __restrict__ topo, const float* __restrict__ cen,
    float* __restrict__ dists) {
  const int lane = threadIdx.x & 31;
  const int wave = threadIdx.x >> 5;
  const int w = blockIdx.x * 8 + wave;     // 0 .. B*T-1
  const int b = w >> 11;
  float4 v = *(const float4*)(topo + (size_t)w * P_ + lane * 4);
  float4 c = *(const float4*)(cen + b * P_ + lane * 4);
  const float dx = v.x - c.x, dy = v.y - c.y, dz = v.z - c.z, dw = v.w - c.w;
  float d = dx * dx + dy * dy + dz * dz + dw * dw;
#pragma unroll
  for (int off = 16; off > 0; off >>= 1) d += __shfl_xor(d, off);
  if (lane == 0) dists[w] = sqrtf(d);
}

// ---------------------------------------------------------------------------
// K4: per-batch bitonic sort (2048 in LDS), kth threshold, masked mean.
// ---------------------------------------------------------------------------
__global__ void __launch_bounds__(256) select_kernel(
    const float* __restrict__ dists, const float* __restrict__ topo,
    const float* __restrict__ imp, float* __restrict__ out) {
  __shared__ float sd[T_];
  __shared__ float cnts[2];
  __shared__ float thr_s;
  const int b = blockIdx.x, tid = threadIdx.x;
  const float* db = dists + (size_t)b * T_;
  for (int i = tid; i < T_; i += 256) sd[i] = db[i];
  __syncthreads();
  for (int k = 2; k <= T_; k <<= 1) {
    for (int j = k >> 1; j > 0; j >>= 1) {
      for (int i = tid; i < T_; i += 256) {
        const int ix = i ^ j;
        if (ix > i) {
          const bool up = ((i & k) == 0);
          const float a = sd[i], c = sd[ix];
          if ((a > c) == up) { sd[i] = c; sd[ix] = a; }
        }
      }
      __syncthreads();
    }
  }
  if (tid == 0) {
    int kk = (int)rintf(imp[b] * 511.f + 1.f);   // round-half-even like jnp
    kk = kk < 1 ? 1 : (kk > 512 ? 512 : kk);
    thr_s = sd[kk - 1];
  }
  __syncthreads();
  const float th = thr_s;
  const int p = tid & 127, half = tid >> 7;
  const float* tb = topo + (size_t)b * T_ * P_;
  float accp = 0.f, cnt = 0.f;
  const int t0 = half * (T_ / 2);
  for (int t = t0; t < t0 + T_ / 2; ++t) {
    const float m = (db[t] <= th) ? 1.f : 0.f;
    accp += m * tb[(size_t)t * P_ + p];
    cnt += m;
  }
  __syncthreads();
  sd[tid] = accp;
  if (p == 0) cnts[half] = cnt;
  __syncthreads();
  if (half == 0) {
    const float tot = sd[tid] + sd[tid + 128];
    const float c = cnts[0] + cnts[1];
    out[b * P_ + p] = tot / fmaxf(c, 1.f);
  }
}

// ---------------------------------------------------------------------------
extern "C" void kernel_launch(void* const* d_in, const int* in_sizes, int n_in,
                              void* d_out, int out_size, void* d_ws, size_t ws_size,
                              hipStream_t stream) {
  (void)in_sizes; (void)n_in; (void)out_size; (void)ws_size;
  const float* mem = (const float*)d_in[0];
  const float* imp = (const float*)d_in[1];
  const float* W1  = (const float*)d_in[2];
  const float* b1  = (const float*)d_in[3];
  const float* W2  = (const float*)d_in[4];
  const float* b2  = (const float*)d_in[5];
  char* ws = (char*)d_ws;
  unsigned short* w1s = (unsigned short*)(ws + OFF_W1S);
  unsigned short* w2s = (unsigned short*)(ws + OFF_W2S);
  float* cen  = (float*)(ws + OFF_CEN);
  float* dst  = (float*)(ws + OFF_DST);
  float* topo = (float*)(ws + OFF_TOPO);
  float* out  = (float*)d_out;

  (void)hipFuncSetAttribute((const void*)topo_map_kernel,
                            hipFuncAttributeMaxDynamicSharedMemorySize, LDS_BYTES);

  stage_weights_kernel<<<72, 256, 0, stream>>>(W1, W2, w1s, w2s);
  topo_map_kernel<<<512, 256, LDS_BYTES, stream>>>(mem, b1, b2, w1s, w2s, topo);
  centroid_kernel<<<B_, 128, 0, stream>>>(topo, cen);
  dists_kernel<<<(B_ * T_) / 8, 256, 0, stream>>>(topo, cen, dst);
  select_kernel<<<B_, 256, 0, stream>>>(dst, topo, imp, out);
}
